// ACCAttention_31061203485393
// MI455X (gfx1250) — compile-verified, matrix operand order checked by reading, not run on hardware
//
#include <hip/hip_runtime.h>
#include <hip/hip_bf16.h>

typedef __bf16 bf16;
typedef __attribute__((ext_vector_type(8)))  bf16  v8bf;
typedef __attribute__((ext_vector_type(16))) bf16  v16bf;
typedef __attribute__((ext_vector_type(8)))  float v8f;

#define HB 96
#define WB 96
#define SPATIAL (HB*WB)      // 9216
#define NBATCH 16
#define QKC 512
#define CVC 256
#define ATT_SCALE 0.08838834764831845f   // 1/sqrt(128)

// ---------- fragment helpers (CDNA5 16-bit WMMA operand layout) ----------
// A/B 16x32 bf16 operand: lane l (half = l>>4) holds K = half*8 + [0..7] and
// K = 16 + half*8 + [0..7]  -> two contiguous 16-byte chunks per lane.
__device__ __forceinline__ v16bf frag16(const bf16* p) {
  v8bf lo = *(const v8bf*)p;
  v8bf hi = *(const v8bf*)(p + 16);
  v16bf r;
#pragma unroll
  for (int i = 0; i < 8; ++i) { r[i] = lo[i]; r[i + 8] = hi[i]; }
  return r;
}

__device__ __forceinline__ v8f vzero8() {
  v8f z = {0.f, 0.f, 0.f, 0.f, 0.f, 0.f, 0.f, 0.f};
  return z;
}

// =======================================================================
// Kernel 1: fused grouped-1x1-conv QKV projection (GEMM, bf16 WMMA)
// Output layouts (channel innermost so attention fragments are contiguous):
//   Q,K : [B][s=h*96+w][512] bf16       V : [B][s][256] bf16
// grid: (144 sTiles, 10 mTiles of 64 rows over [q512|k512|v128-per-group], 32 = b*2+grp)
// =======================================================================
__global__ __launch_bounds__(128) void proj_kernel(
    const float* __restrict__ x,
    const float* __restrict__ wq, const float* __restrict__ bq,
    const float* __restrict__ wk, const float* __restrict__ bk,
    const float* __restrict__ wv, const float* __restrict__ bv,
    bf16* __restrict__ Q, bf16* __restrict__ K, bf16* __restrict__ V)
{
  __shared__ bf16 Wl[64 * 32];   // [oc][ic]   row pitch 32 (64B)
  __shared__ bf16 Xl[64 * 40];   // [s][ic]    row pitch 40 (80B, 16B-mult, conflict-friendly)

  const int sT  = blockIdx.x;
  const int m0  = blockIdx.y * 64;
  const int b   = blockIdx.z >> 1;
  const int grp = blockIdx.z & 1;
  const int s0  = sT * 64;

  const float* wsel; const float* bsel; bf16* dst; int rowBase; int dstC;
  if (m0 < 256)      { wsel = wq; bsel = bq; dst = Q; rowBase = grp * 256 + m0;         dstC = QKC; }
  else if (m0 < 512) { wsel = wk; bsel = bk; dst = K; rowBase = grp * 256 + (m0 - 256); dstC = QKC; }
  else               { wsel = wv; bsel = bv; dst = V; rowBase = grp * 128 + (m0 - 512); dstC = CVC; }

  const float* xg = x + ((long)b * QKC + grp * 256) * SPATIAL;

  const int tid  = threadIdx.x;
  const int wave = tid >> 5, lane = tid & 31, l15 = lane & 15, half = lane >> 4;
  const int mw   = wave * 16;   // this wave's 16-row M strip inside the 64-row tile

  v8f acc[4];
#pragma unroll
  for (int t = 0; t < 4; ++t) acc[t] = vzero8();

  for (int k0 = 0; k0 < 256; k0 += 32) {
    // ---- issue ALL tile loads into registers first (8 b128 loads in flight),
    //      then barrier, then convert+store to LDS ----
    float4 w4[4];
#pragma unroll
    for (int i = 0; i < 4; ++i) {
      int idx = tid + i * 128;          // float4 index, 512 total (64x32 f32)
      int r   = idx >> 3;               // 8 float4 per 32-wide row
      int c4  = (idx & 7) << 2;
      w4[i] = *(const float4*)(wsel + (long)(rowBase + r) * 256 + k0 + c4);
    }
    float4 x4[4];
#pragma unroll
    for (int i = 0; i < 4; ++i) {
      int idx = tid + i * 128;          // float4 index, 512 total (32ic x 64s)
      int ic  = idx >> 4;               // 16 float4 per 64-wide s row
      int s4  = (idx & 15) << 2;
      x4[i] = *(const float4*)(xg + (long)(k0 + ic) * SPATIAL + s0 + s4);
    }
    __syncthreads();   // previous iteration's fragment reads complete
#pragma unroll
    for (int i = 0; i < 4; ++i) {
      int idx = tid + i * 128;
      int r   = idx >> 3;
      int c4  = (idx & 7) << 2;
      Wl[r * 32 + c4 + 0] = (bf16)w4[i].x;
      Wl[r * 32 + c4 + 1] = (bf16)w4[i].y;
      Wl[r * 32 + c4 + 2] = (bf16)w4[i].z;
      Wl[r * 32 + c4 + 3] = (bf16)w4[i].w;
    }
#pragma unroll
    for (int i = 0; i < 4; ++i) {
      int idx = tid + i * 128;
      int ic  = idx >> 4;
      int s4  = (idx & 15) << 2;
      Xl[(s4 + 0) * 40 + ic] = (bf16)x4[i].x;   // transpose to [s][ic]
      Xl[(s4 + 1) * 40 + ic] = (bf16)x4[i].y;
      Xl[(s4 + 2) * 40 + ic] = (bf16)x4[i].z;
      Xl[(s4 + 3) * 40 + ic] = (bf16)x4[i].w;
    }
    __syncthreads();

    v16bf a = frag16(&Wl[(mw + l15) * 32 + half * 8]);
#pragma unroll
    for (int t = 0; t < 4; ++t) {
      v16bf bm = frag16(&Xl[(t * 16 + l15) * 40 + half * 8]);
      acc[t] = __builtin_amdgcn_wmma_f32_16x16x32_bf16(false, a, false, bm,
                                                       (short)0, acc[t], false, false);
    }
  }

  // store: D lane holds column s, VGPR i -> row (oc). Channel-innermost dst ->
  // each lane writes 8 consecutive bf16 channels = one 16B store.
#pragma unroll
  for (int t = 0; t < 4; ++t) {
    int  s     = s0 + t * 16 + l15;
    int  chOff = rowBase + mw + half * 8;
    bf16* op   = dst + ((long)b * SPATIAL + s) * dstC + chOff;
    v8bf o;
#pragma unroll
    for (int i = 0; i < 8; ++i) o[i] = (bf16)(acc[t][i] + bsel[chOff + i]);
    *(v8bf*)op = o;
  }
}

// =======================================================================
// Kernel 2: criss-cross attention for one (batch*head, column) problem.
//   S = Q K^T * scale (96x96, K=128) -> register softmax (half-wave shuffles)
//   O = P V           (96x64,  K=96)
// vertical=0: fix w, attend along h.  vertical=1: fix h, attend along w.
// Writes O to ws layout [bh][col][c(64)][attended(96)] f32.
// block = 192 threads (6 waves, one 16-row strip each). grid = (96, 64).
// =======================================================================
__global__ __launch_bounds__(192) void attn_kernel(
    const bf16* __restrict__ Q, const bf16* __restrict__ K, const bf16* __restrict__ V,
    float* __restrict__ O, int vertical)
{
  // LDS: K tile [96][136] (26112B) aliased after a barrier by P [96][104] (19968B);
  // Vt [64][104] (13312B). Pitches chosen 16B-multiple + conflict-free b128 reads.
  __shared__ __align__(16) unsigned char smem[26112 + 13312];
  bf16* Klds = (bf16*)smem;
  bf16* Plds = (bf16*)smem;              // reuse of K region (after barrier)
  bf16* Vt   = (bf16*)(smem + 26112);

  const int col = blockIdx.x;
  const int bh  = blockIdx.y;
  const int b   = bh >> 2;
  const int head= bh & 3;
  const int tid = threadIdx.x;
  const int wave= tid >> 5, lane = tid & 31, l15 = lane & 15, half = lane >> 4;
  const int m0  = wave * 16;

  const long sBase   = vertical ? (long)col * WB : (long)col;
  const long sStride = vertical ? 1L : (long)WB;

  const bf16* qp = Q + ((long)b * SPATIAL + sBase) * QKC + head * 128;
  const bf16* kp = K + ((long)b * SPATIAL + sBase) * QKC + head * 128;
  const bf16* vp = V + ((long)b * SPATIAL + sBase) * CVC + head * 64;
  const long qRowStride = sStride * QKC;
  const long vRowStride = sStride * CVC;

  // ---- stage K [96 g][128 c] and V^T [64 c][96 g] with 16B vector loads,
  //      all loads issued into registers before the LDS stores ----
  v8bf kbuf[8];                               // 96*128/8 = 1536 chunks / 192 thr
#pragma unroll
  for (int i = 0; i < 8; ++i) {
    int chunk = tid + i * 192;
    int r  = chunk >> 4;                      // 16 chunks per 128-el row
    int c8 = (chunk & 15) << 3;
    kbuf[i] = *(const v8bf*)(kp + (long)r * qRowStride + c8);
  }
  v8bf vbuf[4];                               // 96*64/8 = 768 chunks / 192 thr
#pragma unroll
  for (int i = 0; i < 4; ++i) {
    int chunk = tid + i * 192;
    int g  = chunk >> 3;                      // 8 chunks per 64-el row
    int c8 = (chunk & 7) << 3;
    vbuf[i] = *(const v8bf*)(vp + (long)g * vRowStride + c8);
  }
#pragma unroll
  for (int i = 0; i < 8; ++i) {
    int chunk = tid + i * 192;
    int r  = chunk >> 4;
    int c8 = (chunk & 15) << 3;
    *(v8bf*)&Klds[r * 136 + c8] = kbuf[i];    // ds_store_b128, aligned pitch
  }
#pragma unroll
  for (int i = 0; i < 4; ++i) {
    int chunk = tid + i * 192;
    int g  = chunk >> 3;
    int c8 = (chunk & 7) << 3;
#pragma unroll
    for (int j = 0; j < 8; ++j)
      Vt[(c8 + j) * 104 + g] = vbuf[i][j];    // transpose to [c][g]
  }
  __syncthreads();

  // ---- S = Q K^T : 6 N-tiles per wave, K=128 in 4 steps ----
  v8f accS[6];
#pragma unroll
  for (int t = 0; t < 6; ++t) accS[t] = vzero8();

#pragma unroll
  for (int ks = 0; ks < 128; ks += 32) {
    v16bf a = frag16(qp + (long)(m0 + l15) * qRowStride + ks + half * 8);
#pragma unroll
    for (int t = 0; t < 6; ++t) {
      v16bf bm = frag16(&Klds[(t * 16 + l15) * 136 + ks + half * 8]);
      accS[t] = __builtin_amdgcn_wmma_f32_16x16x32_bf16(false, a, false, bm,
                                                        (short)0, accS[t], false, false);
    }
  }

  // ---- register softmax over the 96 attended positions ----
  // Row r of the C-layout lives entirely in one 16-lane half (lane = N),
  // so 16-wide xor-shuffle reductions (masks 8..1) complete a row.
#pragma unroll
  for (int i = 0; i < 8; ++i) {
    float m = accS[0][i];
#pragma unroll
    for (int t = 1; t < 6; ++t) m = fmaxf(m, accS[t][i]);
#pragma unroll
    for (int msk = 8; msk >= 1; msk >>= 1) m = fmaxf(m, __shfl_xor(m, msk, 32));
    float s = 0.f;
#pragma unroll
    for (int t = 0; t < 6; ++t) {
      float p = __expf((accS[t][i] - m) * ATT_SCALE);
      accS[t][i] = p;
      s += p;
    }
#pragma unroll
    for (int msk = 8; msk >= 1; msk >>= 1) s += __shfl_xor(s, msk, 32);
    float inv = 1.f / s;
#pragma unroll
    for (int t = 0; t < 6; ++t) accS[t][i] *= inv;
  }

  __syncthreads();   // everyone done reading Klds -> safe to overwrite with P
#pragma unroll
  for (int i = 0; i < 8; ++i) {
    int row = m0 + half * 8 + i;
#pragma unroll
    for (int t = 0; t < 6; ++t)
      Plds[row * 104 + t * 16 + l15] = (bf16)accS[t][i];
  }
  __syncthreads();

  // ---- O = P * V : per wave 4 N-tiles (c), K=96 in 3 steps ----
#pragma unroll
  for (int n0 = 0; n0 < 64; n0 += 16) {
    v8f oa = vzero8();
#pragma unroll
    for (int ks = 0; ks < 96; ks += 32) {
      v16bf a  = frag16(&Plds[(m0 + l15) * 104 + ks + half * 8]);
      v16bf bm = frag16(&Vt[(n0 + l15) * 104 + ks + half * 8]);
      oa = __builtin_amdgcn_wmma_f32_16x16x32_bf16(false, a, false, bm,
                                                   (short)0, oa, false, false);
    }
    // lane holds column c; VGPRs i -> 8 consecutive attended rows -> 32B per lane
    float* op = O + (((long)bh * 96 + col) * 64 + (n0 + l15)) * 96 + m0 + half * 8;
    *(float4*)op       = make_float4(oa[0], oa[1], oa[2], oa[3]);
    *((float4*)op + 1) = make_float4(oa[4], oa[5], oa[6], oa[7]);
  }
}

// =======================================================================
// Kernel 3: finalize horizontal part: out[:,0:256] = g*relu(out_h)+x, p_h.
// Oh is [bh][w][c][h]; out is [b][ch][h][w] -> 32x32 LDS transpose tiles.
// grid = (9 tiles, 64 c, 64 bh), 256 threads.
// =======================================================================
__global__ __launch_bounds__(256) void finalize_h_kernel(
    const float* __restrict__ Oh, const float* __restrict__ x,
    const float* __restrict__ gamma,
    const float* __restrict__ bnw, const float* __restrict__ bnb,
    const float* __restrict__ rm,  const float* __restrict__ rv,
    float* __restrict__ out, float* __restrict__ ph)
{
  __shared__ float tile[32][33];
  const int tb = blockIdx.x;
  const int c  = blockIdx.y;
  const int bh = blockIdx.z;
  const int h0 = (tb / 3) * 32, w0 = (tb % 3) * 32;
  const int b = bh >> 2, head = bh & 3;
  const int ch = head * 64 + c;
  const int tx = threadIdx.x & 31, ty = threadIdx.x >> 5;
  const float g = gamma[0];

#pragma unroll
  for (int j = 0; j < 4; ++j) {
    int w = w0 + ty + j * 8;
    tile[ty + j * 8][tx] = Oh[(((long)bh * 96 + w) * 64 + c) * 96 + h0 + tx];
  }
  __syncthreads();

  float inv = 0.f, beta = 0.f;
  if (ch < 128) { inv = bnw[ch] * rsqrtf(rv[ch] + 1e-5f); beta = bnb[ch] - rm[ch] * inv; }

#pragma unroll
  for (int j = 0; j < 4; ++j) {
    int h = h0 + ty + j * 8;
    int w = w0 + tx;
    float v = tile[tx][ty + j * 8];
    long oidx = (((long)b * 512 + ch) * 96 + h) * 96 + w;
    out[oidx] = g * fmaxf(v, 0.f) + x[oidx];
    if (ch < 128) {
      long pidx = (((long)b * 128 + ch) * 96 + h) * 96 + w;
      ph[pidx] = v * inv + beta;
    }
  }
}

// =======================================================================
// Kernel 4: finalize vertical part (Ow layout already matches [h][w]) + p_w.
// grid = (36, 64 c, 64 bh), 256 threads (36*256 = 9216 = H*W).
// =======================================================================
__global__ __launch_bounds__(256) void finalize_w_kernel(
    const float* __restrict__ Ow, const float* __restrict__ x,
    const float* __restrict__ gamma,
    const float* __restrict__ bnw, const float* __restrict__ bnb,
    const float* __restrict__ rm,  const float* __restrict__ rv,
    float* __restrict__ out, float* __restrict__ pw)
{
  const int c  = blockIdx.y;
  const int bh = blockIdx.z;
  const int b = bh >> 2, head = bh & 3;
  const int ch = head * 64 + c;
  const int sidx = blockIdx.x * 256 + threadIdx.x;   // 0..9215
  const int h = sidx / 96;
  const int w = sidx - h * 96;

  float v = Ow[(((long)bh * 96 + h) * 64 + c) * 96 + w];
  const float g = gamma[0];
  long oidx = (((long)b * 512 + 256 + ch) * 96 + h) * 96 + w;
  out[oidx] = g * fmaxf(v, 0.f) + x[oidx];
  if (ch < 128) {
    float inv  = bnw[ch] * rsqrtf(rv[ch] + 1e-5f);
    float beta = bnb[ch] - rm[ch] * inv;
    pw[(((long)b * 128 + ch) * 96 + h) * 96 + w] = v * inv + beta;
  }
}

// =======================================================================
extern "C" void kernel_launch(void* const* d_in, const int* in_sizes, int n_in,
                              void* d_out, int out_size, void* d_ws, size_t ws_size,
                              hipStream_t stream) {
  (void)in_sizes; (void)n_in; (void)out_size; (void)ws_size;

  const float* x     = (const float*)d_in[0];
  const float* wq    = (const float*)d_in[1];
  const float* bq    = (const float*)d_in[2];
  const float* wk    = (const float*)d_in[3];
  const float* bk    = (const float*)d_in[4];
  const float* wv    = (const float*)d_in[5];
  const float* bv    = (const float*)d_in[6];
  const float* bnh_w = (const float*)d_in[7];
  const float* bnh_b = (const float*)d_in[8];
  const float* bnw_w = (const float*)d_in[9];
  const float* bnw_b = (const float*)d_in[10];
  const float* gamma = (const float*)d_in[11];
  const float* bn_rm = (const float*)d_in[12];
  const float* bn_rv = (const float*)d_in[13];

  // workspace layout (bf16 Q/K/V then f32 out_h/out_w)
  const size_t qkElems = (size_t)NBATCH * SPATIAL * QKC;   // 75,497,472
  const size_t vElems  = (size_t)NBATCH * SPATIAL * CVC;   // 37,748,736
  bf16* Qws = (bf16*)d_ws;
  bf16* Kws = Qws + qkElems;
  bf16* Vws = Kws + qkElems;
  float* Oh = (float*)(Vws + vElems);
  float* Ow = Oh + (size_t)64 * 96 * 64 * 96;

  float* outp = (float*)d_out;
  float* php  = outp + (size_t)NBATCH * QKC * SPATIAL;     // out: 75,497,472
  float* pwp  = php  + (size_t)NBATCH * 128 * SPATIAL;     // p_h: 18,874,368

  proj_kernel<<<dim3(144, 10, 32), 128, 0, stream>>>(
      x, wq, bq, wk, bk, wv, bv, Qws, Kws, Vws);

  attn_kernel<<<dim3(96, 64), 192, 0, stream>>>(Qws, Kws, Vws, Oh, 0);
  attn_kernel<<<dim3(96, 64), 192, 0, stream>>>(Qws, Kws, Vws, Ow, 1);

  finalize_h_kernel<<<dim3(9, 64, 64), 256, 0, stream>>>(
      Oh, x, gamma, bnh_w, bnh_b, bn_rm, bn_rv, outp, php);
  finalize_w_kernel<<<dim3(36, 64, 64), 256, 0, stream>>>(
      Ow, x, gamma, bnw_w, bnw_b, bn_rm, bn_rv, outp, pwp);
}
